// Ernie4_5_VLMoeDecoderLayer_39616778338727
// MI455X (gfx1250) — compile-verified, matrix operand order checked by reading, not run on hardware
//
#include <hip/hip_runtime.h>
#include <math.h>
#include <stdint.h>

// ---------------- problem constants ----------------
#define T_TOK   2048
#define D_MODEL 2048
#define NHQ     16
#define NHKV    8
#define HD      128
#define NE      16
#define NTOPK   2
#define FE      1024
#define FSH     2048
#define QKV_W   ((NHQ + 2 * NHKV) * HD)   // 4096

// Async global->LDS (CDNA5 ASYNCcnt path), guarded so absence of the builtin
// falls back to the plain VMEM path instead of failing the build.
#if defined(__has_builtin)
#  if __has_builtin(__builtin_amdgcn_global_load_async_to_lds_b128)
#    define HAVE_ASYNC_LDS 1
#  endif
#endif
#ifndef HAVE_ASYNC_LDS
#  define HAVE_ASYNC_LDS 0
#endif

typedef __attribute__((ext_vector_type(16))) _Float16 v16h;
typedef __attribute__((ext_vector_type(8)))  _Float16 v8h;
typedef __attribute__((ext_vector_type(8)))  float    v8f;

#if HAVE_ASYNC_LDS
typedef int v4i __attribute__((ext_vector_type(4)));
typedef __attribute__((address_space(1))) v4i as1_v4i;  // global int4
typedef __attribute__((address_space(3))) v4i as3_v4i;  // LDS int4
__device__ inline void async_copy16(const void* g, void* l) {
  __builtin_amdgcn_global_load_async_to_lds_b128(
      (as1_v4i*)(uintptr_t)g, (as3_v4i*)(unsigned)(uintptr_t)l, 0, 0);
}
__device__ inline void wait_async0() {
#if __has_builtin(__builtin_amdgcn_s_wait_asynccnt)
  __builtin_amdgcn_s_wait_asynccnt(0);
#else
  asm volatile("s_wait_asynccnt 0x0" ::: "memory");
#endif
}
#endif

union HFrag { v16h v; _Float16 h[16]; };
union FAcc  { v8f  v; float    f[8];  };

__device__ inline v8f wmma_f16(v16h a, v16h b, v8f c) {
  // D = A(16x32 f16) * B(32x16 f16) + C(16x16 f32)
  return __builtin_amdgcn_wmma_f32_16x16x32_f16(false, a, false, b, (short)0, c,
                                                false, false);
}

__device__ inline void zero_acc(FAcc& a) {
  for (int i = 0; i < 8; ++i) a.f[i] = 0.0f;
}

// Fragment = two contiguous 16B runs in LDS (row 16B aligned, pitch multiple
// of 8 f16).  Emits two ds_load_b128 instead of 16x ds_load_u16.
__device__ inline v16h frag_from_lds(const _Float16* row, int lhalf) {
  v8h lo = *(const v8h*)(row + lhalf);
  v8h hi = *(const v8h*)(row + 16 + lhalf);
  return __builtin_shufflevector(lo, hi, 0, 1, 2, 3, 4, 5, 6, 7,
                                 8, 9, 10, 11, 12, 13, 14, 15);
}

// ---------------- RMSNorm ----------------
__global__ __launch_bounds__(256) void rmsnorm_kernel(
    const float* __restrict__ x, const float* __restrict__ w,
    float* __restrict__ out) {
  int t = blockIdx.x;
  const float* row = x + (size_t)t * D_MODEL;
  float ss = 0.0f;
  for (int d = threadIdx.x; d < D_MODEL; d += 256) { float v = row[d]; ss += v * v; }
  __shared__ float red[256];
  red[threadIdx.x] = ss;
  __syncthreads();
  for (int s = 128; s > 0; s >>= 1) {
    if (threadIdx.x < s) red[threadIdx.x] += red[threadIdx.x + s];
    __syncthreads();
  }
  float inv = rsqrtf(red[0] / (float)D_MODEL + 1e-5f);
  for (int d = threadIdx.x; d < D_MODEL; d += 256)
    out[(size_t)t * D_MODEL + d] = row[d] * inv * w[d];
}

// ---------------- generic tiled WMMA GEMM: C = A(MxK) @ B(KxN) [+ addsrc] ----
#define BM 32
#define BN 64
#define BK 32
#define LPITCH 40   // f16 pitch: 80 bytes, keeps 16B alignment, avoids conflicts
#define FPITCH 36   // f32 pitch: 144 bytes, 16B aligned

__global__ __launch_bounds__(256) void wmma_gemm_kernel(
    const float* __restrict__ A, const float* __restrict__ B,
    const float* __restrict__ addsrc, float* __restrict__ C,
    int M, int N, int K) {
#if HAVE_ASYNC_LDS
  __shared__ __align__(16) float sAf[BM][FPITCH];   // fp32, async-staged
#else
  __shared__ __align__(16) _Float16 sA[BM][LPITCH];
#endif
  __shared__ __align__(16) _Float16 sBT[BN][LPITCH]; // B transposed: [n][k]
  int tid  = threadIdx.x;
  int lane = tid & 31;
  int wave = tid >> 5;
  int wm = wave >> 2, wn = wave & 3;   // 2x4 wave grid of 16x16 tiles
  int bm = blockIdx.y * BM, bn = blockIdx.x * BN;
  int l16 = lane & 15;
  int lhalf = (lane >= 16) ? 8 : 0;
  int hi8   = (lane >= 16) ? 8 : 0;

  FAcc acc; zero_acc(acc);
  int arow = wm * 16 + l16;
  int bcol = wn * 16 + l16;

  for (int k0 = 0; k0 < K; k0 += BK) {
    // stage A tile (32x32)
    {
      int r = tid >> 3;
      int c = (tid & 7) * 4;
      const float* gp = A + (size_t)(bm + r) * K + k0 + c;
#if HAVE_ASYNC_LDS
      async_copy16(gp, &sAf[r][c]);       // global_load_async_to_lds_b128
#else
      float4 v = *(const float4*)gp;
      sA[r][c + 0] = (_Float16)v.x; sA[r][c + 1] = (_Float16)v.y;
      sA[r][c + 2] = (_Float16)v.z; sA[r][c + 3] = (_Float16)v.w;
#endif
      if (k0 + BK < K) __builtin_prefetch(gp + BK, 0, 0);
    }
    // stage B tile (32x64) transposed: sBT[n][k]
    {
      int r = tid >> 4;            // k row
      int c = (tid & 15) * 4;      // n col
      for (int rr = r; rr < BK; rr += 16) {
        float4 v = *(const float4*)(B + (size_t)(k0 + rr) * N + bn + c);
        sBT[c + 0][rr] = (_Float16)v.x; sBT[c + 1][rr] = (_Float16)v.y;
        sBT[c + 2][rr] = (_Float16)v.z; sBT[c + 3][rr] = (_Float16)v.w;
      }
    }
#if HAVE_ASYNC_LDS
    wait_async0();                 // s_wait_asynccnt 0 (own copies complete)
#endif
    __syncthreads();
#if HAVE_ASYNC_LDS
    HFrag af;
    {
      const float* rp = &sAf[arow][0];
      float4 x0 = *(const float4*)(rp + lhalf);
      float4 x1 = *(const float4*)(rp + lhalf + 4);
      float4 y0 = *(const float4*)(rp + 16 + lhalf);
      float4 y1 = *(const float4*)(rp + 16 + lhalf + 4);
      af.h[0]  = (_Float16)x0.x; af.h[1]  = (_Float16)x0.y;
      af.h[2]  = (_Float16)x0.z; af.h[3]  = (_Float16)x0.w;
      af.h[4]  = (_Float16)x1.x; af.h[5]  = (_Float16)x1.y;
      af.h[6]  = (_Float16)x1.z; af.h[7]  = (_Float16)x1.w;
      af.h[8]  = (_Float16)y0.x; af.h[9]  = (_Float16)y0.y;
      af.h[10] = (_Float16)y0.z; af.h[11] = (_Float16)y0.w;
      af.h[12] = (_Float16)y1.x; af.h[13] = (_Float16)y1.y;
      af.h[14] = (_Float16)y1.z; af.h[15] = (_Float16)y1.w;
    }
    v16h a = af.v;
#else
    v16h a = frag_from_lds(&sA[arow][0], lhalf);
#endif
    v16h b = frag_from_lds(&sBT[bcol][0], lhalf);
    acc.v = wmma_f16(a, b, acc.v);
    __syncthreads();
  }
  for (int r = 0; r < 8; ++r) {
    int row = bm + wm * 16 + r + hi8;
    int col = bn + bcol;
    size_t idx = (size_t)row * N + col;
    C[idx] = acc.f[r] + (addsrc ? addsrc[idx] : 0.0f);
  }
}

// ---------------- interleaved RoPE with 3-section positions ----------------
__global__ __launch_bounds__(256) void rope_kernel(
    float* __restrict__ qkv, const int* __restrict__ positions) {
  int t = blockIdx.x;
  const int NPAIR = (NHQ + NHKV) * (HD / 2);   // rotate q and k heads
  for (int p = threadIdx.x; p < NPAIR; p += 256) {
    int head = p >> 6;          // 0..23
    int j    = p & 63;          // freq index within head
    int sec  = (j < 22) ? 0 : ((j < 44) ? 1 : 2);
    float pos = (float)positions[sec * T_TOK + t];
    float inv = __powf(500000.0f, -((float)(2 * j) / (float)HD));
    float ang = pos * inv;
    float c = __cosf(ang), s = __sinf(ang);
    int off = head * HD;        // q heads 0..15, k heads follow immediately
    float* ptr = qkv + (size_t)t * QKV_W + off + 2 * j;
    float x1 = ptr[0], x2 = ptr[1];
    ptr[0] = x1 * c - x2 * s;
    ptr[1] = x2 * c + x1 * s;
  }
}

// ---------------- flash attention, causal GQA, one wave per (head, qtile) ----
__global__ __launch_bounds__(32) void attn_kernel(
    const float* __restrict__ qkv, float* __restrict__ out) {
  int h  = blockIdx.x;
  int q0 = blockIdx.y * 16;
  int lane  = threadIdx.x & 31;
  int l16   = lane & 15;
  int lhalf = (lane >= 16) ? 8 : 0;
  int hi8   = (lane >= 16) ? 8 : 0;
  int kh = h >> 1;                       // GQA: 2 q heads per kv head
  const float scale = 0.08838834764831845f;  // 1/sqrt(128)

  // Q fragments: 4 hd-tiles of K=32, scale folded in; explicit b128 loads
  HFrag qf[4];
  {
    const float* qp = qkv + (size_t)(q0 + l16) * QKV_W + h * HD;
    for (int kt = 0; kt < 4; ++kt) {
      int base = kt * 32 + lhalf;
      float4 a0 = *(const float4*)(qp + base);
      float4 a1 = *(const float4*)(qp + base + 4);
      float4 b0 = *(const float4*)(qp + base + 16);
      float4 b1 = *(const float4*)(qp + base + 20);
      qf[kt].h[0]  = (_Float16)(a0.x * scale); qf[kt].h[1]  = (_Float16)(a0.y * scale);
      qf[kt].h[2]  = (_Float16)(a0.z * scale); qf[kt].h[3]  = (_Float16)(a0.w * scale);
      qf[kt].h[4]  = (_Float16)(a1.x * scale); qf[kt].h[5]  = (_Float16)(a1.y * scale);
      qf[kt].h[6]  = (_Float16)(a1.z * scale); qf[kt].h[7]  = (_Float16)(a1.w * scale);
      qf[kt].h[8]  = (_Float16)(b0.x * scale); qf[kt].h[9]  = (_Float16)(b0.y * scale);
      qf[kt].h[10] = (_Float16)(b0.z * scale); qf[kt].h[11] = (_Float16)(b0.w * scale);
      qf[kt].h[12] = (_Float16)(b1.x * scale); qf[kt].h[13] = (_Float16)(b1.y * scale);
      qf[kt].h[14] = (_Float16)(b1.z * scale); qf[kt].h[15] = (_Float16)(b1.w * scale);
    }
  }

  FAcc o[8];
  for (int j = 0; j < 8; ++j) zero_acc(o[j]);
  float m_run[8], l_run[8];
  for (int r = 0; r < 8; ++r) { m_run[r] = -1e30f; l_run[r] = 0.0f; }

  __shared__ __align__(16) _Float16 sP[16][LPITCH];

  int kend = q0 + 16;
  for (int kb = 0; kb < kend; kb += 32) {
    // ---- scores: Q @ K^T for two 16-key subtiles ----
    FAcc s[2];
    for (int kt2 = 0; kt2 < 2; ++kt2) {
      zero_acc(s[kt2]);
      int key = kb + kt2 * 16 + l16;
      const float* kp = qkv + (size_t)key * QKV_W + NHQ * HD + kh * HD;
      for (int hdt = 0; hdt < 4; ++hdt) {
        HFrag kf;
        int base = hdt * 32 + lhalf;
        float4 a0 = *(const float4*)(kp + base);
        float4 a1 = *(const float4*)(kp + base + 4);
        float4 b0 = *(const float4*)(kp + base + 16);
        float4 b1 = *(const float4*)(kp + base + 20);
        kf.h[0]  = (_Float16)a0.x; kf.h[1]  = (_Float16)a0.y;
        kf.h[2]  = (_Float16)a0.z; kf.h[3]  = (_Float16)a0.w;
        kf.h[4]  = (_Float16)a1.x; kf.h[5]  = (_Float16)a1.y;
        kf.h[6]  = (_Float16)a1.z; kf.h[7]  = (_Float16)a1.w;
        kf.h[8]  = (_Float16)b0.x; kf.h[9]  = (_Float16)b0.y;
        kf.h[10] = (_Float16)b0.z; kf.h[11] = (_Float16)b0.w;
        kf.h[12] = (_Float16)b1.x; kf.h[13] = (_Float16)b1.y;
        kf.h[14] = (_Float16)b1.z; kf.h[15] = (_Float16)b1.w;
        s[kt2].v = wmma_f16(qf[hdt].v, kf.v, s[kt2].v);
      }
    }
    // ---- causal mask ----
    for (int kt2 = 0; kt2 < 2; ++kt2)
      for (int r = 0; r < 8; ++r) {
        int qrow = q0 + r + hi8;
        int key  = kb + kt2 * 16 + l16;
        if (key > qrow) s[kt2].f[r] = -3.0e38f;
      }
    // ---- online softmax (row reductions across half-wave lanes) ----
    float sc[8];
    for (int r = 0; r < 8; ++r) {
      float v = fmaxf(s[0].f[r], s[1].f[r]);
      for (int off = 1; off < 16; off <<= 1) v = fmaxf(v, __shfl_xor(v, off, 32));
      float mnew = fmaxf(m_run[r], v);
      sc[r] = __expf(m_run[r] - mnew);
      m_run[r] = mnew;
    }
    for (int kt2 = 0; kt2 < 2; ++kt2)
      for (int r = 0; r < 8; ++r)
        s[kt2].f[r] = __expf(s[kt2].f[r] - m_run[r]);
    for (int r = 0; r < 8; ++r) {
      float v = s[0].f[r] + s[1].f[r];
      for (int off = 1; off < 16; off <<= 1) v += __shfl_xor(v, off, 32);
      l_run[r] = l_run[r] * sc[r] + v;
      for (int j = 0; j < 8; ++j) o[j].f[r] *= sc[r];
    }
    // ---- transpose P (C layout -> A layout) through LDS ----
    for (int kt2 = 0; kt2 < 2; ++kt2)
      for (int r = 0; r < 8; ++r)
        sP[r + hi8][kt2 * 16 + l16] = (_Float16)s[kt2].f[r];
    __syncthreads();
    v16h pa = frag_from_lds(&sP[l16][0], lhalf);
    // ---- PV: P(16x32) @ V(32x128) in 8 n-tiles ----
    for (int j = 0; j < 8; ++j) {
      const float* vp = qkv + (NHQ + NHKV) * HD + kh * HD + j * 16 + l16;
      HFrag vf;
      for (int i = 0; i < 8; ++i) {
        vf.h[i]     = (_Float16)vp[(size_t)(kb + lhalf + i) * QKV_W];
        vf.h[8 + i] = (_Float16)vp[(size_t)(kb + 16 + lhalf + i) * QKV_W];
      }
      o[j].v = wmma_f16(pa, vf.v, o[j].v);
    }
    __syncthreads();
  }
  // ---- epilogue ----
  for (int r = 0; r < 8; ++r) {
    float invl = 1.0f / l_run[r];
    int row = q0 + r + hi8;
    for (int j = 0; j < 8; ++j)
      out[(size_t)row * (NHQ * HD) + h * HD + j * 16 + l16] = o[j].f[r] * invl;
  }
}

// ---------------- shared-expert SwiGLU elementwise ----------------
__global__ __launch_bounds__(256) void silu_mul_kernel(
    const float* __restrict__ g, const float* __restrict__ u,
    float* __restrict__ out, int n) {
  int i = blockIdx.x * 256 + threadIdx.x;
  if (i < n) { float x = g[i]; out[i] = (x / (1.0f + __expf(-x))) * u[i]; }
}

// ---------------- MoE gate: softmax + top-2 routing, one wave per token ----
__global__ __launch_bounds__(32) void gate_topk_kernel(
    const float* __restrict__ h, const float* __restrict__ gw,
    const float* __restrict__ gb, int* __restrict__ plist,
    int* __restrict__ counts, float* __restrict__ expw) {
  int t = blockIdx.x;
  int lane = threadIdx.x & 31;
  const float* hr = h + (size_t)t * D_MODEL;
  float logits[NE];
  for (int e = 0; e < NE; ++e) {
    float p = 0.0f;
    for (int d = lane; d < D_MODEL; d += 32) p += hr[d] * gw[(size_t)d * NE + e];
    for (int off = 1; off < 32; off <<= 1) p += __shfl_xor(p, off, 32);
    logits[e] = p;
  }
  float mx = -1e30f;
  for (int e = 0; e < NE; ++e) mx = fmaxf(mx, logits[e]);
  float pr[NE], sum = 0.0f;
  for (int e = 0; e < NE; ++e) { pr[e] = __expf(logits[e] - mx); sum += pr[e]; }
  float inv = 1.0f / sum;
  for (int e = 0; e < NE; ++e) pr[e] *= inv;
  int i0 = 0, i1 = 1;
  float b0 = -1e30f, b1 = -1e30f;
  for (int e = 0; e < NE; ++e) {
    float s = pr[e] + gb[e];
    if (s > b0) { b1 = b0; i1 = i0; b0 = s; i0 = e; }
    else if (s > b1) { b1 = s; i1 = e; }
  }
  float w0 = pr[i0], w1 = pr[i1];
  float wsum = w0 + w1;
  w0 /= wsum; w1 /= wsum;
  if (lane == 0) {
    int p0 = atomicAdd(&counts[i0], 1); plist[i0 * T_TOK + p0] = t * 2 + 0;
    int p1 = atomicAdd(&counts[i1], 1); plist[i1 * T_TOK + p1] = t * 2 + 1;
    expw[t * 2 + 0] = w0; expw[t * 2 + 1] = w1;
  }
}

// ---------------- MoE up-projection: act = silu(h@w1[e]) * (h@w3[e]) * cw ----
__global__ __launch_bounds__(256) void moe_up_kernel(
    const float* __restrict__ h, const float* __restrict__ w1,
    const float* __restrict__ w3, const int* __restrict__ plist,
    const int* __restrict__ counts, const float* __restrict__ expw,
    float* __restrict__ act) {
  int e = blockIdx.x;
  int cnt = counts[e];
  int tb = blockIdx.y * 16;
  if (tb >= cnt) return;
  int fb = blockIdx.z * 128;
  int nval = (cnt - tb < 16) ? (cnt - tb) : 16;

  __shared__ int   sPair[16];
  __shared__ float sCw[16];
  __shared__ __align__(16) _Float16 sA[16][LPITCH];
  __shared__ __align__(16) _Float16 sB1T[128][LPITCH];
  __shared__ __align__(16) _Float16 sB3T[128][LPITCH];

  int tid = threadIdx.x, lane = tid & 31, wave = tid >> 5;
  int l16 = lane & 15;
  int lhalf = (lane >= 16) ? 8 : 0;
  int hi8   = (lane >= 16) ? 8 : 0;
  if (tid < 16) {
    int pr = (tid < nval) ? plist[e * T_TOK + tb + tid] : 0;
    sPair[tid] = pr;
    sCw[tid]   = (tid < nval) ? expw[pr] : 0.0f;
  }
  __syncthreads();

  FAcc g, u; zero_acc(g); zero_acc(u);
  const float* w1e = w1 + (size_t)e * D_MODEL * FE;
  const float* w3e = w3 + (size_t)e * D_MODEL * FE;
  int bcol = wave * 16 + l16;     // 0..127

  for (int k0 = 0; k0 < D_MODEL; k0 += 32) {
    {   // gathered activation rows (16x32), row-major
      int r = tid >> 4;
      int c = (tid & 15) * 2;
      float2 v = make_float2(0.0f, 0.0f);
      if (r < nval) v = *(const float2*)(h + (size_t)(sPair[r] >> 1) * D_MODEL + k0 + c);
      sA[r][c] = (_Float16)v.x; sA[r][c + 1] = (_Float16)v.y;
    }
    {   // weight tiles (32x128) for w1/w3, stored transposed: sB*T[f][k]
      int r = tid >> 3;            // k row 0..31
      int c = (tid & 7) * 16;      // f col
      const float* p1 = w1e + (size_t)(k0 + r) * FE + fb + c;
      const float* p3 = w3e + (size_t)(k0 + r) * FE + fb + c;
      for (int i = 0; i < 16; i += 4) {
        float4 v1 = *(const float4*)(p1 + i);
        sB1T[c + i + 0][r] = (_Float16)v1.x; sB1T[c + i + 1][r] = (_Float16)v1.y;
        sB1T[c + i + 2][r] = (_Float16)v1.z; sB1T[c + i + 3][r] = (_Float16)v1.w;
        float4 v3 = *(const float4*)(p3 + i);
        sB3T[c + i + 0][r] = (_Float16)v3.x; sB3T[c + i + 1][r] = (_Float16)v3.y;
        sB3T[c + i + 2][r] = (_Float16)v3.z; sB3T[c + i + 3][r] = (_Float16)v3.w;
      }
    }
    __syncthreads();
    v16h a  = frag_from_lds(&sA[l16][0], lhalf);
    v16h b1 = frag_from_lds(&sB1T[bcol][0], lhalf);
    g.v = wmma_f16(a, b1, g.v);
    v16h b3 = frag_from_lds(&sB3T[bcol][0], lhalf);
    u.v = wmma_f16(a, b3, u.v);
    __syncthreads();
  }
  for (int r = 0; r < 8; ++r) {
    int row = r + hi8;
    if (row < nval) {
      float gv = g.f[r];
      float val = (gv / (1.0f + __expf(-gv))) * u.f[r] * sCw[row];
      act[(size_t)sPair[row] * FE + fb + wave * 16 + l16] = val;
    }
  }
}

// ---------------- MoE down-projection: moe_part[slot][t] = act @ w2[e] ------
__global__ __launch_bounds__(256) void moe_down_kernel(
    const float* __restrict__ act, const float* __restrict__ w2,
    const int* __restrict__ plist, const int* __restrict__ counts,
    float* __restrict__ moe_part) {
  int e = blockIdx.x;
  int cnt = counts[e];
  int tb = blockIdx.y * 16;
  if (tb >= cnt) return;
  int db = blockIdx.z * 128;
  int nval = (cnt - tb < 16) ? (cnt - tb) : 16;

  __shared__ int sPair[16];
  __shared__ __align__(16) _Float16 sA[16][LPITCH];
  __shared__ __align__(16) _Float16 sBT[128][LPITCH];

  int tid = threadIdx.x, lane = tid & 31, wave = tid >> 5;
  int l16 = lane & 15;
  int lhalf = (lane >= 16) ? 8 : 0;
  int hi8   = (lane >= 16) ? 8 : 0;
  if (tid < 16) sPair[tid] = (tid < nval) ? plist[e * T_TOK + tb + tid] : 0;
  __syncthreads();

  FAcc acc; zero_acc(acc);
  const float* w2e = w2 + (size_t)e * FE * D_MODEL;
  int bcol = wave * 16 + l16;     // 0..127

  for (int k0 = 0; k0 < FE; k0 += 32) {
    {
      int r = tid >> 4;
      int c = (tid & 15) * 2;
      float2 v = make_float2(0.0f, 0.0f);
      if (r < nval) v = *(const float2*)(act + (size_t)sPair[r] * FE + k0 + c);
      sA[r][c] = (_Float16)v.x; sA[r][c + 1] = (_Float16)v.y;
    }
    {
      int r = tid >> 3;
      int c = (tid & 7) * 16;
      const float* p = w2e + (size_t)(k0 + r) * D_MODEL + db + c;
      for (int i = 0; i < 16; i += 4) {
        float4 v = *(const float4*)(p + i);
        sBT[c + i + 0][r] = (_Float16)v.x; sBT[c + i + 1][r] = (_Float16)v.y;
        sBT[c + i + 2][r] = (_Float16)v.z; sBT[c + i + 3][r] = (_Float16)v.w;
      }
    }
    __syncthreads();
    v16h a = frag_from_lds(&sA[l16][0], lhalf);
    v16h b = frag_from_lds(&sBT[bcol][0], lhalf);
    acc.v = wmma_f16(a, b, acc.v);
    __syncthreads();
  }
  for (int r = 0; r < 8; ++r) {
    int row = r + hi8;
    if (row < nval) {
      int pair = sPair[row];
      int slot = pair & 1;
      int tok  = pair >> 1;
      moe_part[(size_t)slot * T_TOK * D_MODEL + (size_t)tok * D_MODEL +
               db + wave * 16 + l16] = acc.f[r];
    }
  }
}

// ---------------- final combine: out += moe_slot0 + moe_slot1 ---------------
__global__ __launch_bounds__(256) void final_add_kernel(
    float* __restrict__ out, const float* __restrict__ mp) {
  size_t i = (size_t)blockIdx.x * 256 + threadIdx.x;
  out[i] += mp[i] + mp[(size_t)T_TOK * D_MODEL + i];
}

// ---------------- launcher ----------------
extern "C" void kernel_launch(void* const* d_in, const int* in_sizes, int n_in,
                              void* d_out, int out_size, void* d_ws, size_t ws_size,
                              hipStream_t stream) {
  (void)in_sizes; (void)n_in; (void)out_size; (void)ws_size;
  const int*   positions = (const int*)d_in[0];
  const float* hidden    = (const float*)d_in[1];
  // d_in[2]: visual_token_mask — unused by the reference computation
  const float* w_norm1 = (const float*)d_in[3];
  const float* w_norm2 = (const float*)d_in[4];
  const float* wqkv    = (const float*)d_in[5];
  const float* wo      = (const float*)d_in[6];
  const float* gate_w  = (const float*)d_in[7];
  const float* gate_b  = (const float*)d_in[8];
  const float* w1      = (const float*)d_in[9];
  const float* w3      = (const float*)d_in[10];
  const float* w2      = (const float*)d_in[11];
  const float* ws1     = (const float*)d_in[12];
  const float* ws3     = (const float*)d_in[13];
  const float* ws2     = (const float*)d_in[14];
  float* out = (float*)d_out;

  float* ws = (float*)d_ws;
  size_t off = 0;
  auto alloc = [&](size_t n) { float* p = ws + off; off += n; return p; };
  float* x1     = alloc((size_t)T_TOK * D_MODEL);
  float* qkv    = alloc((size_t)T_TOK * QKV_W);
  float* attn   = alloc((size_t)T_TOK * NHQ * HD);
  float* resid2 = alloc((size_t)T_TOK * D_MODEL);
  float* h2     = alloc((size_t)T_TOK * D_MODEL);
  float* gsh    = alloc((size_t)T_TOK * FSH);
  float* ush    = alloc((size_t)T_TOK * FSH);
  float* actb   = alloc((size_t)T_TOK * NTOPK * FE);
  float* mpart  = alloc((size_t)2 * T_TOK * D_MODEL);
  float* expw   = alloc((size_t)T_TOK * 2);
  int*   plist  = (int*)alloc((size_t)NE * T_TOK);
  int*   counts = (int*)alloc(64);

  (void)hipMemsetAsync(counts, 0, NE * sizeof(int), stream);

  // attention block
  rmsnorm_kernel<<<T_TOK, 256, 0, stream>>>(hidden, w_norm1, x1);
  wmma_gemm_kernel<<<dim3(QKV_W / BN, T_TOK / BM), 256, 0, stream>>>(
      x1, wqkv, nullptr, qkv, T_TOK, QKV_W, D_MODEL);
  rope_kernel<<<T_TOK, 256, 0, stream>>>(qkv, positions);
  attn_kernel<<<dim3(NHQ, T_TOK / 16), 32, 0, stream>>>(qkv, attn);
  wmma_gemm_kernel<<<dim3(D_MODEL / BN, T_TOK / BM), 256, 0, stream>>>(
      attn, wo, hidden, resid2, T_TOK, D_MODEL, NHQ * HD);

  // shared-expert MLP
  rmsnorm_kernel<<<T_TOK, 256, 0, stream>>>(resid2, w_norm2, h2);
  wmma_gemm_kernel<<<dim3(FSH / BN, T_TOK / BM), 256, 0, stream>>>(
      h2, ws1, nullptr, gsh, T_TOK, FSH, D_MODEL);
  wmma_gemm_kernel<<<dim3(FSH / BN, T_TOK / BM), 256, 0, stream>>>(
      h2, ws3, nullptr, ush, T_TOK, FSH, D_MODEL);
  silu_mul_kernel<<<(T_TOK * FSH + 255) / 256, 256, 0, stream>>>(
      gsh, ush, gsh, T_TOK * FSH);
  wmma_gemm_kernel<<<dim3(D_MODEL / BN, T_TOK / BM), 256, 0, stream>>>(
      gsh, ws2, resid2, out, T_TOK, D_MODEL, FSH);

  // routed MoE (top-2 sparse)
  gate_topk_kernel<<<T_TOK, 32, 0, stream>>>(h2, gate_w, gate_b, plist, counts, expw);
  moe_up_kernel<<<dim3(NE, T_TOK / 16, FE / 128), 256, 0, stream>>>(
      h2, w1, w3, plist, counts, expw, actb);
  moe_down_kernel<<<dim3(NE, T_TOK / 16, D_MODEL / 128), 256, 0, stream>>>(
      actb, w2, plist, counts, mpart);
  final_add_kernel<<<(T_TOK * D_MODEL) / 256, 256, 0, stream>>>(out, mpart);
}
